// HybridGPT_11768210391696
// MI455X (gfx1250) — matrix operand order repaired and checked by reading, not compiled, not run on hardware
//
#include <hip/hip_runtime.h>
#include <hip/hip_bf16.h>
#include <math.h>

// ---------------------------------------------------------------------------
// HybridGPT forward for MI455X (gfx1250, wave32, WMMA).
// Weight-bandwidth-bound: fp32 weights streamed once via b128 loads, converted
// in-register to bf16 (v_cvt_pk_bf16_f32), multiplied with
// v_wmma_f32_16x16x32_bf16 (fp32 accum), double-buffered LDS pipeline.
// ---------------------------------------------------------------------------

#define TT   1024
#define DD   2048
#define VV   16384
#define NH   16
#define HDIM 128
#define DINN 4096
#define DST  16
#define DTRK 128
#define MLPH 8192
#define WINW 128

#define BSTR 40  // padded B-tile LDS row stride (elements); 80B keeps b128 aligned

typedef __attribute__((ext_vector_type(16))) __bf16 v16bf;
typedef __attribute__((ext_vector_type(8)))  __bf16 v8bf;
typedef __attribute__((ext_vector_type(4)))  __bf16 v4bf;
typedef __attribute__((ext_vector_type(8)))  float  v8f;

// ------------------------------ reductions ---------------------------------
__device__ __forceinline__ float wave_sum32(float v) {
#pragma unroll
  for (int off = 16; off > 0; off >>= 1) v += __shfl_xor(v, off, 32);
  return v;
}

// block-wide sum, result broadcast to all threads. red[] must have >= nwaves.
__device__ __forceinline__ float block_sum(float v, float* red) {
  v = wave_sum32(v);
  const int nw = blockDim.x >> 5;
  if ((threadIdx.x & 31) == 0) red[threadIdx.x >> 5] = v;
  __syncthreads();
  float s = 0.f;
  for (int i = 0; i < nw; i++) s += red[i];
  __syncthreads();
  return s;
}

// ------------------------- WMMA bf16 GEMM ----------------------------------
// C[M,N](fp32) = epi( A[M,K](fp32,lda) x W[K,N](fp32,ldw) )
// tile 64x64 per 128-thread block (4 waves); K-step 32; double-buffered LDS.
// Requires: K % 32 == 0, N % 4 == 0, M % 64 == 0 (all call sites satisfy).
// EPI: 0 store, 1 add (residual), 2 relu^2, 3 bias+softplus.
template <int EPI>
__global__ __launch_bounds__(128)
void gemm_wmma(const float* __restrict__ A, int lda,
               const float* __restrict__ W, int ldw,
               float* __restrict__ C, int ldc,
               const float* __restrict__ bias,
               int M, int N, int K) {
  __shared__ __bf16 As[2][64 * 32];    // [m][k]
  __shared__ __bf16 Bs[2][64 * BSTR];  // [n][k] transposed, padded stride

  const int tid   = threadIdx.x;
  const int lane  = tid & 31;
  const int wv    = tid >> 5;  // wave -> 16-row M sub-tile
  const int mbase = blockIdx.y * 64;
  const int nbase = blockIdx.x * 64;

  v8f acc[4];
#pragma unroll
  for (int j = 0; j < 4; j++)
#pragma unroll
    for (int r = 0; r < 8; r++) acc[j][r] = 0.0f;

  const int hi16  = (lane >= 16);
  const int arow  = (wv << 4) + (lane & 15);
  const int akoff = hi16 ? 8 : 0;   // ISA 16-bit A 16x32 layout
  const int bkoff = hi16 ? 8 : 0;   // B K order matches A: {0..7,16..23} / {8..15,24..31}

  // ---- staging: 4 explicit passes, float4 global loads (b128), bf16 pack ----
  float4 ra[4], rb[4];
  auto stage_load = [&](int k0) {
#pragma unroll
    for (int p = 0; p < 4; p++) {
      const int ea = (tid + p * 128) << 2;      // A element idx, 4-aligned
      const int am = ea >> 5, ak = ea & 31;
      ra[p] = *(const float4*)&A[(size_t)(mbase + am) * lda + (k0 + ak)];
      const int eb = (tid + p * 128) << 2;      // B element idx (k-major read)
      const int bk = eb >> 6, bn = eb & 63;
      if (nbase + bn < N)
        rb[p] = *(const float4*)&W[(size_t)(k0 + bk) * ldw + (nbase + bn)];
      else
        rb[p] = make_float4(0.f, 0.f, 0.f, 0.f);
    }
  };
  auto stage_store = [&](int buf) {
#pragma unroll
    for (int p = 0; p < 4; p++) {
      const int ea = (tid + p * 128) << 2;
      v4bf ta;
      ta[0] = (__bf16)ra[p].x; ta[1] = (__bf16)ra[p].y;
      ta[2] = (__bf16)ra[p].z; ta[3] = (__bf16)ra[p].w;
      *(v4bf*)&As[buf][ea] = ta;                // b64 LDS store
      const int eb = (tid + p * 128) << 2;
      const int bk = eb >> 6, bn = eb & 63;
      Bs[buf][(bn + 0) * BSTR + bk] = (__bf16)rb[p].x;
      Bs[buf][(bn + 1) * BSTR + bk] = (__bf16)rb[p].y;
      Bs[buf][(bn + 2) * BSTR + bk] = (__bf16)rb[p].z;
      Bs[buf][(bn + 3) * BSTR + bk] = (__bf16)rb[p].w;
    }
  };

  stage_load(0);
  stage_store(0);
  __syncthreads();

  int cur = 0;
  for (int k0 = 0; k0 < K; k0 += 32) {
    const bool more = (k0 + 32) < K;
    if (more) stage_load(k0 + 32);  // HBM loads in flight during WMMA below

    // A fragment: lane<16 rows M=l, K {0..7,16..23}; lane>=16 K {8..15,24..31}
    const v8bf alo = *(const v8bf*)&As[cur][arow * 32 + akoff];
    const v8bf ahi = *(const v8bf*)&As[cur][arow * 32 + akoff + 16];
    v16bf af;
#pragma unroll
    for (int i = 0; i < 8; i++) { af[i] = alo[i]; af[i + 8] = ahi[i]; }

#pragma unroll
    for (int j = 0; j < 4; j++) {
      const int bcol = (j << 4) + (lane & 15);
      const v8bf blo = *(const v8bf*)&Bs[cur][bcol * BSTR + bkoff];
      const v8bf bhi = *(const v8bf*)&Bs[cur][bcol * BSTR + bkoff + 16];
      v16bf bfrag;
#pragma unroll
      for (int i = 0; i < 8; i++) { bfrag[i] = blo[i]; bfrag[i + 8] = bhi[i]; }
      acc[j] = __builtin_amdgcn_wmma_f32_16x16x32_bf16(
          false, af, false, bfrag, (short)0, acc[j], false, false);
    }

    if (more) stage_store(cur ^ 1);
    __syncthreads();
    cur ^= 1;
  }

  // epilogue: C layout lanes0-15 M=r, lanes16-31 M=8+r; N=lane&15
  const int mrow0 = mbase + (wv << 4) + (hi16 ? 8 : 0);
#pragma unroll
  for (int j = 0; j < 4; j++) {
    const int n = nbase + (j << 4) + (lane & 15);
    if (n >= N) continue;
#pragma unroll
    for (int r = 0; r < 8; r++) {
      const int m = mrow0 + r;
      if (m >= M) continue;
      float v = acc[j][r];
      const size_t off = (size_t)m * ldc + n;
      if (EPI == 0) C[off] = v;
      else if (EPI == 1) C[off] += v;
      else if (EPI == 2) { const float t = v > 0.f ? v : 0.f; C[off] = t * t; }
      else { v += bias[n]; C[off] = (v > 20.f) ? v : log1pf(__expf(v)); }
    }
  }
}

// ------------------------- elementwise / norm kernels ----------------------
__global__ void rope_tables_kernel(float* __restrict__ cosb, float* __restrict__ sinb) {
  const int i = blockIdx.x * blockDim.x + threadIdx.x;
  if (i >= TT * 64) return;
  const int t = i >> 6, j = i & 63;
  const float inv = powf(10000.0f, -((float)(2 * j)) / 128.0f);
  const float a = (float)t * inv;
  cosb[i] = cosf(a); sinb[i] = sinf(a);
}

__global__ __launch_bounds__(256)
void embed_rmsnorm_kernel(const float* __restrict__ wte, const int* __restrict__ idx,
                          float* __restrict__ x, float* __restrict__ x0) {
  __shared__ float red[8];
  const int t = blockIdx.x;
  const float* row = wte + (size_t)idx[t] * DD;
  float ss = 0.f;
  for (int d = threadIdx.x; d < DD; d += blockDim.x) { const float v = row[d]; ss += v * v; }
  ss = block_sum(ss, red);
  const float inv = rsqrtf(ss / (float)DD + 1e-6f);
  for (int d = threadIdx.x; d < DD; d += blockDim.x) {
    const float v = row[d] * inv;
    x[(size_t)t * DD + d]  = v;
    x0[(size_t)t * DD + d] = v;
  }
}

__global__ __launch_bounds__(256)
void rmsnorm_kernel(const float* __restrict__ in, float* __restrict__ out, int Dn) {
  __shared__ float red[8];
  const int t = blockIdx.x;
  const float* p = in + (size_t)t * Dn;
  float ss = 0.f;
  for (int d = threadIdx.x; d < Dn; d += blockDim.x) { const float v = p[d]; ss += v * v; }
  ss = block_sum(ss, red);
  const float inv = rsqrtf(ss / (float)Dn + 1e-6f);
  for (int d = threadIdx.x; d < Dn; d += blockDim.x) out[(size_t)t * Dn + d] = p[d] * inv;
}

__global__ void resid_kernel(float* __restrict__ x, const float* __restrict__ x0,
                             const float* __restrict__ rl, const float* __restrict__ x0l,
                             int bi, int n) {
  const int i = blockIdx.x * blockDim.x + threadIdx.x;
  if (i < n) x[i] = rl[bi] * x[i] + x0l[bi] * x0[i];
}

// rope + rmsnorm per (t, head), 128 threads (HD dims)
__global__ __launch_bounds__(128)
void rope_rms_kernel(float* __restrict__ buf, const float* __restrict__ cosb,
                     const float* __restrict__ sinb) {
  __shared__ float red[4];
  const int t = blockIdx.x, hh = blockIdx.y, d = threadIdx.x;
  float* p = buf + (size_t)t * 2048 + hh * HDIM;
  const int j = d & 63;
  const float x1 = p[j], x2 = p[j + 64];
  const float c = cosb[t * 64 + j], s = sinb[t * 64 + j];
  const float v = (d < 64) ? (x1 * c + x2 * s) : (-x1 * s + x2 * c);
  const float ss = block_sum(v * v, red);  // barrier also orders read->write
  p[d] = v * rsqrtf(ss / (float)HDIM + 1e-6f);
}

// v += 2*sigmoid(h[:, :32] @ Wg) * ve[idx]
__global__ __launch_bounds__(256)
void gate_ve_kernel(const float* __restrict__ h, const float* __restrict__ Wg,
                    const float* __restrict__ ve, const int* __restrict__ idx,
                    float* __restrict__ v) {
  __shared__ float g[NH];
  const int t = blockIdx.x;
  if (threadIdx.x < NH) {
    float s = 0.f;
    for (int c = 0; c < 32; c++) s += h[(size_t)t * DD + c] * Wg[c * NH + threadIdx.x];
    g[threadIdx.x] = 2.0f / (1.0f + __expf(-s));
  }
  __syncthreads();
  const float* vr = ve + (size_t)idx[t] * 2048;
  for (int e = threadIdx.x; e < 2048; e += blockDim.x)
    v[(size_t)t * 2048 + e] += g[e >> 7] * vr[e];
}

// windowed causal attention, online softmax; one block per (t, head)
__global__ __launch_bounds__(128)
void attn_kernel(const float* __restrict__ Q, const float* __restrict__ Kb,
                 const float* __restrict__ Vb, float* __restrict__ Y) {
  __shared__ float red[4];
  const int t = blockIdx.x, hh = blockIdx.y, d = threadIdx.x;
  const float scale = 0.08838834764831845f;  // 1/sqrt(128)
  const float qd = Q[(size_t)t * 2048 + hh * HDIM + d];
  int s0 = t - WINW; if (s0 < 0) s0 = 0;
  float m = -3.4e38f, l = 0.f, acc = 0.f;
  for (int s = s0; s <= t; s++) {
    const float kd  = Kb[(size_t)s * 2048 + hh * HDIM + d];
    const float sco = block_sum(qd * kd, red) * scale;
    const float mn   = fmaxf(m, sco);
    const float corr = __expf(m - mn);
    const float p    = __expf(sco - mn);
    l   = l * corr + p;
    acc = acc * corr + p * Vb[(size_t)s * 2048 + hh * HDIM + d];
    m = mn;
  }
  Y[(size_t)t * 2048 + hh * HDIM + d] = acc / l;
}

// depthwise causal conv (DCONV=4) + silu; x part of xz (row stride 8192)
__global__ void conv_silu_kernel(const float* __restrict__ xz, const float* __restrict__ cw,
                                 const float* __restrict__ cb, float* __restrict__ xc) {
  const int i = blockIdx.x * blockDim.x + threadIdx.x;
  if (i >= TT * DINN) return;
  const int t = i >> 12, d = i & (DINN - 1);
  float s = cb[d];
#pragma unroll
  for (int j = 0; j < 4; j++) {
    const int ts = t - 3 + j;
    if (ts >= 0) s += xz[(size_t)ts * 8192 + d] * cw[d * 4 + j];
  }
  xc[i] = s / (1.0f + __expf(-s));
}

// selective scan: thread per (d, n); 16-lane xor-reduce over states
__global__ __launch_bounds__(256)
void scan_kernel(const float* __restrict__ dt, const float* __restrict__ xc,
                 const float* __restrict__ dbc, const float* __restrict__ Alog,
                 const float* __restrict__ Dp, float* __restrict__ y) {
  const int g = blockIdx.x * 256 + threadIdx.x;  // DINN*DST threads
  const int d = g >> 4, n = g & 15;
  const float A  = -__expf(Alog[d * DST + n]);
  const float Dd = Dp[d];
  float st = 0.f;
  for (int t = 0; t < TT; t++) {
    const float dtv = dt[(size_t)t * DINN + d];
    const float xv  = xc[(size_t)t * DINN + d];
    const float Bv  = dbc[t * 160 + 128 + n];
    const float Cv  = dbc[t * 160 + 144 + n];
    st = __expf(dtv * A) * st + dtv * Bv * xv;
    float part = st * Cv;
#pragma unroll
    for (int off = 8; off > 0; off >>= 1) part += __shfl_xor(part, off, 16);
    if (n == 0) y[(size_t)t * DINN + d] = part + Dd * xv;
  }
}

__global__ void gated_mul_kernel(float* __restrict__ y, const float* __restrict__ xz) {
  const int i = blockIdx.x * blockDim.x + threadIdx.x;
  if (i >= TT * DINN) return;
  const int t = i >> 12, d = i & (DINN - 1);
  const float z = xz[(size_t)t * 8192 + 4096 + d];
  y[i] *= z / (1.0f + __expf(-z));
}

// ------------------------------- host side ---------------------------------
static void launch_gemm(int epi, const float* A, int lda, const float* W, int ldw,
                        float* C, int ldc, const float* bias,
                        int M, int N, int K, hipStream_t s) {
  dim3 grid((N + 63) / 64, (M + 63) / 64), block(128);
  switch (epi) {
    case 0: gemm_wmma<0><<<grid, block, 0, s>>>(A, lda, W, ldw, C, ldc, bias, M, N, K); break;
    case 1: gemm_wmma<1><<<grid, block, 0, s>>>(A, lda, W, ldw, C, ldc, bias, M, N, K); break;
    case 2: gemm_wmma<2><<<grid, block, 0, s>>>(A, lda, W, ldw, C, ldc, bias, M, N, K); break;
    default: gemm_wmma<3><<<grid, block, 0, s>>>(A, lda, W, ldw, C, ldc, bias, M, N, K); break;
  }
}

extern "C" void kernel_launch(void* const* d_in, const int* in_sizes, int n_in,
                              void* d_out, int out_size, void* d_ws, size_t ws_size,
                              hipStream_t stream) {
  (void)in_sizes; (void)n_in; (void)out_size; (void)ws_size;

  // inputs in setup_inputs() dict order (nested dicts depth-first, insertion order)
  const int*   idx = (const int*)d_in[0];
  const float* wte = (const float*)d_in[1];
  const float* lmh = (const float*)d_in[2];
  const float* rl  = (const float*)d_in[3];
  const float* x0l = (const float*)d_in[4];
  // attn0: Wq Wk Wv Wo Wg @5..9 ; attn2 @10..14
  const float* aWq[2] = {(const float*)d_in[5],  (const float*)d_in[10]};
  const float* aWk[2] = {(const float*)d_in[6],  (const float*)d_in[11]};
  const float* aWv[2] = {(const float*)d_in[7],  (const float*)d_in[12]};
  const float* aWo[2] = {(const float*)d_in[8],  (const float*)d_in[13]};
  const float* aWg[2] = {(const float*)d_in[9],  (const float*)d_in[14]};
  const float* veT[2] = {(const float*)d_in[15], (const float*)d_in[16]};
  const float* m_in   = (const float*)d_in[17];  // in_proj  [D, 2*DIN]
  const float* m_cw   = (const float*)d_in[18];  // conv_w   [DIN, 4]
  const float* m_cb   = (const float*)d_in[19];  // conv_b
  const float* m_xp   = (const float*)d_in[20];  // x_proj   [DIN, 160]
  const float* m_dtw  = (const float*)d_in[21];  // dt_w     [128, DIN]
  const float* m_dtb  = (const float*)d_in[22];  // dt_b
  const float* m_Al   = (const float*)d_in[23];  // A_log    [DIN, 16]
  const float* m_D    = (const float*)d_in[24];  // D
  const float* m_op   = (const float*)d_in[25];  // out_proj [DIN, D]
  const float* mW1[3] = {(const float*)d_in[26], (const float*)d_in[28], (const float*)d_in[30]};
  const float* mW2[3] = {(const float*)d_in[27], (const float*)d_in[29], (const float*)d_in[31]};
  float* out = (float*)d_out;

  // workspace layout (floats)
  float* ws = (float*)d_ws;
  size_t o = 0;
  float* x    = ws + o; o += (size_t)TT * DD;
  float* x0b  = ws + o; o += (size_t)TT * DD;
  float* hb   = ws + o; o += (size_t)TT * DD;
  float* pA   = ws + o; o += (size_t)TT * 8192;  // mlp hidden / mamba xz
  float* pB   = ws + o; o += (size_t)TT * 8192;  // attn q,k,v,y | mamba xc,dt
  float* ymb  = ws + o; o += (size_t)TT * DINN;
  float* dbcb = ws + o; o += (size_t)TT * 160;
  float* cosb = ws + o; o += (size_t)TT * 64;
  float* sinb = ws + o; o += (size_t)TT * 64;

  float* qb = pB;
  float* kb = pB + (size_t)TT * 2048;
  float* vb = pB + (size_t)2 * TT * 2048;
  float* yb = pB + (size_t)3 * TT * 2048;
  float* xcb = pB;                      // aliases attn bufs (disjoint lifetime)
  float* dtb = pB + (size_t)TT * DINN;

  const int TD = TT * DD;
  const int TDIN = TT * DINN;

  rope_tables_kernel<<<(TT * 64 + 255) / 256, 256, 0, stream>>>(cosb, sinb);
  embed_rmsnorm_kernel<<<TT, 256, 0, stream>>>(wte, idx, x, x0b);

  const int kinds[3] = {0, 1, 0};  // attn, mamba, attn
  int ai = 0;
  for (int bi = 0; bi < 3; bi++) {
    resid_kernel<<<(TD + 255) / 256, 256, 0, stream>>>(x, x0b, rl, x0l, bi, TD);
    rmsnorm_kernel<<<TT, 256, 0, stream>>>(x, hb, DD);

    if (kinds[bi] == 0) {
      // --- attention block ---
      launch_gemm(0, hb, DD, aWq[ai], 2048, qb, 2048, nullptr, TT, 2048, DD, stream);
      launch_gemm(0, hb, DD, aWk[ai], 2048, kb, 2048, nullptr, TT, 2048, DD, stream);
      launch_gemm(0, hb, DD, aWv[ai], 2048, vb, 2048, nullptr, TT, 2048, DD, stream);
      gate_ve_kernel<<<TT, 256, 0, stream>>>(hb, aWg[ai], veT[ai], idx, vb);
      dim3 gh(TT, NH);
      rope_rms_kernel<<<gh, 128, 0, stream>>>(qb, cosb, sinb);
      rope_rms_kernel<<<gh, 128, 0, stream>>>(kb, cosb, sinb);
      attn_kernel<<<gh, 128, 0, stream>>>(qb, kb, vb, yb);
      launch_gemm(1, yb, 2048, aWo[ai], DD, x, DD, nullptr, TT, DD, 2048, stream);
      ai++;
    } else {
      // --- mamba block ---
      launch_gemm(0, hb, DD, m_in, 2 * DINN, pA, 2 * DINN, nullptr, TT, 2 * DINN, DD, stream);
      conv_silu_kernel<<<(TDIN + 255) / 256, 256, 0, stream>>>(pA, m_cw, m_cb, xcb);
      launch_gemm(0, xcb, DINN, m_xp, 160, dbcb, 160, nullptr, TT, 160, DINN, stream);
      launch_gemm(3, dbcb, 160, m_dtw, DINN, dtb, DINN, m_dtb, TT, DINN, DTRK, stream);
      scan_kernel<<<(DINN * DST) / 256, 256, 0, stream>>>(dtb, xcb, dbcb, m_Al, m_D, ymb);
      gated_mul_kernel<<<(TDIN + 255) / 256, 256, 0, stream>>>(ymb, pA);
      launch_gemm(1, ymb, DINN, m_op, DD, x, DD, nullptr, TT, DD, DINN, stream);
    }

    // --- MLP: x += relu(h@W1)^2 @ W2 ---
    rmsnorm_kernel<<<TT, 256, 0, stream>>>(x, hb, DD);
    launch_gemm(2, hb, DD, mW1[bi], MLPH, pA, MLPH, nullptr, TT, MLPH, DD, stream);
    launch_gemm(1, pA, MLPH, mW2[bi], DD, x, DD, nullptr, TT, DD, MLPH, stream);
  }

  // final rmsnorm + lm_head
  rmsnorm_kernel<<<TT, 256, 0, stream>>>(x, hb, DD);
  launch_gemm(0, hb, DD, lmh, VV, out, VV, nullptr, TT, VV, DD, stream);
}
